// ContModel_72103910965340
// MI455X (gfx1250) — matrix operand order repaired and checked by reading, not compiled, not run on hardware
//
#include <hip/hip_runtime.h>
#include <math.h>

typedef __attribute__((ext_vector_type(4)))  float v4f;
typedef __attribute__((ext_vector_type(8)))  float v8f;
typedef __attribute__((ext_vector_type(8)))  __bf16 v8bf;
typedef __attribute__((ext_vector_type(16))) __bf16 v16bf;
typedef __attribute__((ext_vector_type(4)))  int   v4i;
typedef unsigned short u16;
typedef unsigned int   u32;
typedef __attribute__((ext_vector_type(4))) unsigned short u16x4;
typedef __attribute__((ext_vector_type(4))) unsigned int   u32x4;

#define NUM_CLASS 1000
#define PAD_CLASS 1024
#define LOW_DIM   128
#define B_UPD     1024
#define B_SIM     65536
#define PROTO_M   0.99f

#if defined(__gfx1250__) && __has_builtin(__builtin_amdgcn_global_load_async_to_lds_b128)
#define HAVE_ASYNC_LDS 1
#endif

// ---- bf16 split helpers (x = hi + lo, both bf16 rtne) ----------------------
__device__ __forceinline__ u16 bf16_rtne(float x) {
    union { float f; u32 u; } v; v.f = x;
    u32 r = v.u + 0x7FFFu + ((v.u >> 16) & 1u);
    return (u16)(r >> 16);
}
__device__ __forceinline__ float bf16_f32(u16 h) {
    union { u32 u; float f; } v; v.u = ((u32)h) << 16;
    return v.f;
}
__device__ __forceinline__ void split2(float x, u16 &hi, u16 &lo) {
    hi = bf16_rtne(x);
    lo = bf16_rtne(x - bf16_f32(hi));
}

// ---------------------------------------------------------------------------
// Kernel 1: per-class sequential EMA (== the lax.scan recurrence, since row c
// depends only on samples with label==c, in batch order) + L2 normalize.
// Emits hi/lo bf16 prototypes, zero-padded to 1024 classes.
// ---------------------------------------------------------------------------
__global__ __launch_bounds__(128)
void proto_ema_norm_split(const float* __restrict__ prototypes,
                          const float* __restrict__ pred_feat,
                          const int*   __restrict__ pseudo_label,
                          u16* __restrict__ protos_hi,
                          u16* __restrict__ protos_lo)
{
    __shared__ int   labs[B_UPD];
    __shared__ float red[LOW_DIM];
    const int c = blockIdx.x;
    const int t = threadIdx.x;

    if (c >= NUM_CLASS) {               // zero padding rows 1000..1023
        protos_hi[(size_t)c * LOW_DIM + t] = 0;
        protos_lo[(size_t)c * LOW_DIM + t] = 0;
        return;
    }

    for (int i = t; i < B_UPD; i += 128) labs[i] = pseudo_label[i];
    __syncthreads();

    float p = prototypes[(size_t)c * LOW_DIM + t];
    for (int i = 0; i < B_UPD; ++i) {
        if (labs[i] == c) {             // block-uniform branch -> scalar skip
            p = fmaf(PROTO_M, p, (1.0f - PROTO_M) * pred_feat[(size_t)i * LOW_DIM + t]);
        }
    }

    red[t] = p * p;
    __syncthreads();
    for (int s = 64; s > 0; s >>= 1) {
        if (t < s) red[t] += red[t + s];
        __syncthreads();
    }
    const float scale = 1.0f / fmaxf(sqrtf(red[0]), 1e-12f);

    u16 hi, lo;
    split2(p * scale, hi, lo);
    protos_hi[(size_t)c * LOW_DIM + t] = hi;
    protos_lo[(size_t)c * LOW_DIM + t] = lo;
}

// ---------------------------------------------------------------------------
// Kernel 2: stream-split feat (fp32) into hi/lo bf16 arrays. 65 MB traffic.
// ---------------------------------------------------------------------------
__global__ __launch_bounds__(256)
void feat_split(const float* __restrict__ x,
                u16* __restrict__ xh, u16* __restrict__ xl)
{
    const size_t i = ((size_t)blockIdx.x * 256 + threadIdx.x) * 4;
    v4f f = *(const v4f*)(x + i);
    u16x4 hh, ll;
#pragma unroll
    for (int j = 0; j < 4; ++j) {
        u16 a, b; split2(f[j], a, b);
        hh[j] = a; ll[j] = b;
    }
    *(u16x4*)(xh + i) = hh;
    *(u16x4*)(xl + i) = ll;
}

// ---------------------------------------------------------------------------
// Kernel 3: sim = feat @ protos^T with V_WMMA_F32_16X16X32_BF16, split-bf16:
//   acc += Ahi*Bhi + Alo*Bhi + Ahi*Blo   (f32 accumulate, ~fp32 accuracy)
// Block = 256 threads (8 wave32); 64-class bf16 hi/lo B panel in LDS
// (row stride 144 bf16 = 288 B); each wave: one 16-row M tile, whole K=128
// A panel (hi+lo) in 64 VGPRs, 4 N-subtiles x 12 WMMAs.
// ---------------------------------------------------------------------------
#define BN   64
#define LDBE 144

__global__ __launch_bounds__(256)
void sim_gemm_bf16(const u16* __restrict__ feat_hi, const u16* __restrict__ feat_lo,
                   const u16* __restrict__ protos_hi, const u16* __restrict__ protos_lo,
                   float* __restrict__ out)
{
    __shared__ u16 BsH[BN * LDBE];   // 18 KB
    __shared__ u16 BsL[BN * LDBE];   // 18 KB
    const int mblk = blockIdx.x;     // 0..511
    const int nblk = blockIdx.y;     // 0..15
    const int tid  = threadIdx.x;

    // ---- stage 64x128 bf16 hi/lo panels as 16B chunks --------------------
#ifdef HAVE_ASYNC_LDS
    for (int c = tid; c < BN * 16; c += 256) {
        const int r  = c >> 4;
        const int k8 = (c & 15) * 8;
        const size_t g = (size_t)(nblk * BN + r) * LOW_DIM + k8;
        __builtin_amdgcn_global_load_async_to_lds_b128(
            (__attribute__((address_space(1))) v4i*)(protos_hi + g),
            (__attribute__((address_space(3))) v4i*)&BsH[r * LDBE + k8], 0, 0);
        __builtin_amdgcn_global_load_async_to_lds_b128(
            (__attribute__((address_space(1))) v4i*)(protos_lo + g),
            (__attribute__((address_space(3))) v4i*)&BsL[r * LDBE + k8], 0, 0);
    }
    asm volatile("s_wait_asynccnt 0x0" ::: "memory");
#else
    for (int c = tid; c < BN * 16; c += 256) {
        const int r  = c >> 4;
        const int k8 = (c & 15) * 8;
        const size_t g = (size_t)(nblk * BN + r) * LOW_DIM + k8;
        *(u32x4*)&BsH[r * LDBE + k8] = *(const u32x4*)(protos_hi + g);
        *(u32x4*)&BsL[r * LDBE + k8] = *(const u32x4*)(protos_lo + g);
    }
#endif
    __syncthreads();

    const int wave = tid >> 5;
    const int lane = tid & 31;
    const int h    = lane >> 4;   // K-half select per ISA 16-bit A/B layouts
    const int l16  = lane & 15;

    const int row0 = (mblk * 8 + wave) * 16;
    const u16* ah = feat_hi + (size_t)(row0 + l16) * LOW_DIM;
    const u16* al = feat_lo + (size_t)(row0 + l16) * LOW_DIM;

    // Whole 16x128 A panel, hi+lo: 8 x v16bf = 64 VGPRs.
    v16bf Ahi[4], Alo[4];
#pragma unroll
    for (int s = 0; s < 4; ++s) {
        v8bf h0 = *(const v8bf*)(ah + 32 * s + 8 * h);        // K = 32s+8h..+7
        v8bf h1 = *(const v8bf*)(ah + 32 * s + 16 + 8 * h);   // K = 32s+16+8h..+7
        Ahi[s] = __builtin_shufflevector(h0, h1, 0,1,2,3,4,5,6,7,8,9,10,11,12,13,14,15);
        v8bf l0 = *(const v8bf*)(al + 32 * s + 8 * h);
        v8bf l1 = *(const v8bf*)(al + 32 * s + 16 + 8 * h);
        Alo[s] = __builtin_shufflevector(l0, l1, 0,1,2,3,4,5,6,7,8,9,10,11,12,13,14,15);
    }

#pragma unroll
    for (int nt = 0; nt < 4; ++nt) {
        const int colbase = nt * 16;
        const u16* bh = &BsH[(colbase + l16) * LDBE + 8 * h];
        const u16* bl = &BsL[(colbase + l16) * LDBE + 8 * h];

        v8f acc = {0.f, 0.f, 0.f, 0.f, 0.f, 0.f, 0.f, 0.f};
#pragma unroll
        for (int s = 0; s < 4; ++s) {
            v8bf b0 = *(const v8bf*)(bh + 32 * s);
            v8bf b1 = *(const v8bf*)(bh + 32 * s + 16);
            v16bf Bhi = __builtin_shufflevector(b0, b1, 0,1,2,3,4,5,6,7,8,9,10,11,12,13,14,15);
            v8bf c0 = *(const v8bf*)(bl + 32 * s);
            v8bf c1 = *(const v8bf*)(bl + 32 * s + 16);
            v16bf Blo = __builtin_shufflevector(c0, c1, 0,1,2,3,4,5,6,7,8,9,10,11,12,13,14,15);

            acc = __builtin_amdgcn_wmma_f32_16x16x32_bf16(false, Ahi[s], false, Bhi,
                                                          (short)0, acc, false, false);
            acc = __builtin_amdgcn_wmma_f32_16x16x32_bf16(false, Alo[s], false, Bhi,
                                                          (short)0, acc, false, false);
            acc = __builtin_amdgcn_wmma_f32_16x16x32_bf16(false, Ahi[s], false, Blo,
                                                          (short)0, acc, false, false);
        }

        const int col = nblk * BN + colbase + l16;
        if (col < NUM_CLASS) {
            float* op = out + (size_t)(row0 + 8 * h) * NUM_CLASS + col;
#pragma unroll
            for (int r = 0; r < 8; ++r)
                op[(size_t)r * NUM_CLASS] = acc[r];
        }
    }
}

// ---------------------------------------------------------------------------
extern "C" void kernel_launch(void* const* d_in, const int* in_sizes, int n_in,
                              void* d_out, int out_size, void* d_ws, size_t ws_size,
                              hipStream_t stream)
{
    const float* pred_feat    = (const float*)d_in[0];
    const int*   pseudo_label = (const int*)d_in[1];
    const float* prototypes   = (const float*)d_in[2];
    const float* feat         = (const float*)d_in[3];
    float*       out          = (float*)d_out;

    // Workspace layout (all 16B aligned): ~32.5 MB total.
    u16* protos_hi = (u16*)d_ws;                                  // 1024*128
    u16* protos_lo = protos_hi + (size_t)PAD_CLASS * LOW_DIM;     // 1024*128
    u16* feat_hi   = protos_lo + (size_t)PAD_CLASS * LOW_DIM;     // 65536*128
    u16* feat_lo   = feat_hi   + (size_t)B_SIM * LOW_DIM;         // 65536*128

    hipLaunchKernelGGL(proto_ema_norm_split, dim3(PAD_CLASS), dim3(128), 0, stream,
                       prototypes, pred_feat, pseudo_label, protos_hi, protos_lo);

    hipLaunchKernelGGL(feat_split, dim3((B_SIM * LOW_DIM) / (256 * 4)), dim3(256), 0, stream,
                       feat, feat_hi, feat_lo);

    hipLaunchKernelGGL(sim_gemm_bf16,
                       dim3(B_SIM / (8 * 16), PAD_CLASS / BN), dim3(256), 0, stream,
                       feat_hi, feat_lo, protos_hi, protos_lo, out);
}
